// CharRNN_86792699118085
// MI455X (gfx1250) — compile-verified, matrix operand order checked by reading, not run on hardware
//
#include <hip/hip_runtime.h>
#include <math.h>

// CharRNN on MI455X (gfx1250, wave32, WMMA).
// 8 persistent WGs; weights + h staged in LDS via global_load_async_to_lds;
// bf16 WMMA GEMMs; device-wide barrier per timestep.

#define B_    64
#define L_    1024
#define EMB_  128
#define HID_  512
#define OUT_  256
#define NWG   8

// padded LDS row strides (elements): +8 ushort = +4 dwords -> bank-conflict free,
// 16B alignment preserved.
#define HPAD  520
#define EPAD  136

typedef __attribute__((ext_vector_type(16))) __bf16 v16bf;
typedef __attribute__((ext_vector_type(8)))  float  v8f;

union FragAB { v16bf h; unsigned int u[8]; };

// ---- workspace layout (bytes, all 256-aligned) ----
#define OFF_BAR 0                         // 256 B barrier counter area
#define OFF_H   256                       // 2 * 64*512 bf16  = 131072
#define OFF_WET (OFF_H   + 131072)        // 512*128 bf16 (W_e^T, N-major) = 131072
#define OFF_WHT (OFF_WET + 131072)        // 512*512 bf16 (W_h^T)          = 524288
#define OFF_WOT (OFF_WHT + 524288)        // 256*512 bf16 (W_o^T)          = 262144
#define OFF_EMB (OFF_WOT + 262144)        // 256*128 bf16 (embedding)      = 65536

__device__ __forceinline__ unsigned short f2bf(float f) {
  unsigned int u = __float_as_uint(f);
  u += 0x7fffu + ((u >> 16) & 1u);        // round-to-nearest-even
  return (unsigned short)(u >> 16);
}

// Async 16B global -> LDS copy (per-lane). VDST operand = LDS byte offset
// (low 32 bits of the flat shared address); tracked by ASYNCcnt.
__device__ __forceinline__ void async_ld16(unsigned short* dst_lds,
                                           const unsigned short* src) {
  unsigned lds_off = (unsigned)(size_t)(void*)dst_lds;
  unsigned long long ga = (unsigned long long)(size_t)(const void*)src;
  asm volatile("global_load_async_to_lds_b128 %0, %1, off"
               :: "v"(lds_off), "v"(ga) : "memory");
}

__device__ __forceinline__ void async_wait_all() {
  asm volatile("s_wait_asynccnt 0x0" ::: "memory");
}

// 16x32 bf16 A/B fragment load from a row pointer (row-major, 16B-aligned rows).
// ISA layout: lanes 0-15 row=lane&15, K pairs {0..7,16..23}; lanes 16-31: +8.
__device__ __forceinline__ v16bf load_frag_row(const unsigned short* rowp, int k0, int lane) {
  const unsigned int* q = (const unsigned int*)(rowp + k0);
  const int kb = (lane < 16) ? 0 : 8;
  FragAB f;
#pragma unroll
  for (int g = 0; g < 8; ++g) {
    const int kh = (g < 4) ? (kb + 2 * g) : (16 + kb + 2 * (g - 4));
    f.u[g] = q[kh >> 1];
  }
  return f.h;
}

__device__ __forceinline__ v8f wmma_bf16(v16bf a, v16bf b, v8f c) {
  return __builtin_amdgcn_wmma_f32_16x16x32_bf16(false, a, false, b, (short)0, c,
                                                 false, false);
}

// Device-wide barrier across NWG persistent workgroups (monotonic counter).
__device__ __forceinline__ void grid_barrier(unsigned int* bar, unsigned int target) {
  __syncthreads();
  if (threadIdx.x == 0) {
    __threadfence();                       // release h writes to device scope
    atomicAdd(bar, 1u);
    while (__hip_atomic_load(bar, __ATOMIC_RELAXED, __HIP_MEMORY_SCOPE_AGENT) < target) {
      __builtin_amdgcn_s_sleep(1);
    }
  }
  __syncthreads();
  __threadfence();                         // acquire other WGs' h writes
}

// ---- prep: transpose + f32->bf16 convert weights, init h0 ----
__global__ __launch_bounds__(256) void charrnn_prep(
    const float* __restrict__ hidden, const float* __restrict__ embedding,
    const float* __restrict__ W_e, const float* __restrict__ W_h,
    const float* __restrict__ W_o, unsigned char* __restrict__ ws) {
  const int idx = blockIdx.x * blockDim.x + threadIdx.x;
  unsigned short* wet  = (unsigned short*)(ws + OFF_WET);
  unsigned short* wht  = (unsigned short*)(ws + OFF_WHT);
  unsigned short* wot  = (unsigned short*)(ws + OFF_WOT);
  unsigned short* embb = (unsigned short*)(ws + OFF_EMB);
  unsigned short* h0   = (unsigned short*)(ws + OFF_H);

  if (idx < 65536) {                               // W_e^T [512][128]
    const int n = idx >> 7, k = idx & 127;
    wet[idx] = f2bf(W_e[k * HID_ + n]);
  } else if (idx < 65536 + 262144) {               // W_h^T [512][512]
    const int j = idx - 65536, n = j >> 9, k = j & 511;
    wht[j] = f2bf(W_h[k * HID_ + n]);
  } else if (idx < 65536 + 262144 + 131072) {      // W_o^T [256][512]
    const int j = idx - (65536 + 262144), n = j >> 9, k = j & 511;
    wot[j] = f2bf(W_o[k * OUT_ + n]);
  } else if (idx < 65536 + 262144 + 131072 + 32768) {  // embedding bf16
    const int j = idx - (65536 + 262144 + 131072);
    embb[j] = f2bf(embedding[j]);
  } else if (idx < 65536 + 262144 + 131072 + 65536) {  // h0 <- hidden
    const int j = idx - (65536 + 262144 + 131072 + 32768);
    h0[j] = f2bf(hidden[j]);
  }
}

// ---- persistent recurrent kernel: 8 WGs x 256 threads (8 waves) ----
__global__ __launch_bounds__(256, 1) void charrnn_recurrent(
    const int* __restrict__ x, const float* __restrict__ b_h,
    const float* __restrict__ b_o, float* __restrict__ out,
    unsigned char* __restrict__ ws) {
  // LDS staging (padded rows): ~180 KB of the 320 KB WGP budget.
  __shared__ __align__(16) unsigned short sh_h [B_ * HPAD];   // 66560 B
  __shared__ __align__(16) unsigned short sh_wh[64 * HPAD];   // 66560 B
  __shared__ __align__(16) unsigned short sh_wo[32 * HPAD];   // 33280 B
  __shared__ __align__(16) unsigned short sh_we[64 * EPAD];   // 17408 B

  unsigned int*         bar  = (unsigned int*)(ws + OFF_BAR);
  unsigned short*       hbuf = (unsigned short*)(ws + OFF_H);
  const unsigned short* wet  = (const unsigned short*)(ws + OFF_WET);
  const unsigned short* wht  = (const unsigned short*)(ws + OFF_WHT);
  const unsigned short* wot  = (const unsigned short*)(ws + OFF_WOT);
  const unsigned short* embb = (const unsigned short*)(ws + OFF_EMB);

  const int w    = blockIdx.x;        // 0..7: owns h cols [64w,64w+64), logit cols [32w,32w+32)
  const int wv   = threadIdx.x >> 5;  // wave 0..7
  const int lane = threadIdx.x & 31;
  const int tid  = threadIdx.x;

  // ---- one-time: stage this WG's weight slices into LDS (async) ----
  for (int i = tid; i < 4096; i += 256) {             // W_h^T slice: 64 rows x 512
    const int row = i >> 6, c8 = (i & 63) * 8;
    async_ld16(&sh_wh[row * HPAD + c8], wht + (w * 64 + row) * HID_ + c8);
  }
  for (int i = tid; i < 2048; i += 256) {             // W_o^T slice: 32 rows x 512
    const int row = i >> 6, c8 = (i & 63) * 8;
    async_ld16(&sh_wo[row * HPAD + c8], wot + (w * 32 + row) * HID_ + c8);
  }
  for (int i = tid; i < 1024; i += 256) {             // W_e^T slice: 64 rows x 128
    const int row = i >> 4, c8 = (i & 15) * 8;
    async_ld16(&sh_we[row * EPAD + c8], wet + (w * 64 + row) * EMB_ + c8);
  }
  async_wait_all();
  __syncthreads();

  // h-GEMM: 4x4 tiles of 16x16; wave wv -> tiles (mt, nt0) and (mt, nt0+1)
  const int mt  = wv >> 1;
  const int m0  = mt * 16;
  const int nt0 = (2 * wv) & 3;       // 0 or 2
  const int nc0 = w * 64 + nt0 * 16;  // global h columns (for bias + global stores)
  const int nc1 = nc0 + 16;

  // logits-GEMM: 4x2 tiles; wave wv -> tile (wv>>1, wv&1)
  const int lm0   = (wv >> 1) * 16;
  const int lncol = w * 32 + (wv & 1) * 16;

  const int nidx = lane & 15;
  const int rofs = (lane < 16) ? 0 : 8;   // C/D: lanes 16-31 hold M+8

  const float bias_h0 = b_h[nc0 + nidx];
  const float bias_h1 = b_h[nc1 + nidx];
  const float bias_o  = b_o[lncol + nidx];

  const int arow  = m0 + nidx;        // this lane's A row (batch index) for h-GEMM
  const int larow = lm0 + nidx;       // A row for logits-GEMM

  // per-lane B row pointers in LDS
  const unsigned short* be0 = &sh_we[(nt0 * 16 + nidx) * EPAD];
  const unsigned short* be1 = &sh_we[((nt0 + 1) * 16 + nidx) * EPAD];
  const unsigned short* bh0 = &sh_wh[(nt0 * 16 + nidx) * HPAD];
  const unsigned short* bh1 = &sh_wh[((nt0 + 1) * 16 + nidx) * HPAD];
  const unsigned short* bo  = &sh_wo[((wv & 1) * 16 + nidx) * HPAD];
  const unsigned short* ahrow_l = &sh_h[arow * HPAD];
  const unsigned short* alrow_l = &sh_h[larow * HPAD];

  const size_t FINAL_OFS = (size_t)B_ * L_ * OUT_;

  for (int t = 0; t < L_; ++t) {
    // h_{t-1} lives in hbuf[t&1]; we write h_t into hbuf[(t+1)&1]
    const unsigned short* hcur = hbuf + (size_t)(t & 1) * (B_ * HID_);
    unsigned short*       hnxt = hbuf + (size_t)((t + 1) & 1) * (B_ * HID_);

    // ---- stage h_{t-1} into LDS (async, 64 KB) ----
    for (int i = tid; i < 4096; i += 256) {
      const int row = i >> 6, c8 = (i & 63) * 8;
      async_ld16(&sh_h[row * HPAD + c8], hcur + row * HID_ + c8);
    }
    async_wait_all();
    __syncthreads();

    // ---- h_t = tanh(E_t*W_e + h_{t-1}*W_h + b_h), cols [nc0, nc0+32) ----
    v8f acc0, acc1;
#pragma unroll
    for (int r = 0; r < 8; ++r) { acc0[r] = bias_h0; acc1[r] = bias_h1; }

    {   // embedding contribution, K = 128 (A rows gathered from global by token)
      const int tok = x[arow * L_ + t];
      const unsigned short* aerow = embb + tok * EMB_;
#pragma unroll
      for (int kc = 0; kc < EMB_; kc += 32) {
        const v16bf a = load_frag_row(aerow, kc, lane);
        acc0 = wmma_bf16(a, load_frag_row(be0, kc, lane), acc0);
        acc1 = wmma_bf16(a, load_frag_row(be1, kc, lane), acc1);
      }
    }
    {   // recurrent contribution, K = 512 (A and B from LDS)
#pragma unroll 4
      for (int kc = 0; kc < HID_; kc += 32) {
        const v16bf a = load_frag_row(ahrow_l, kc, lane);
        acc0 = wmma_bf16(a, load_frag_row(bh0, kc, lane), acc0);
        acc1 = wmma_bf16(a, load_frag_row(bh1, kc, lane), acc1);
      }
    }
    // tanh, publish bf16 h_t to global (and f32 final hidden at last step)
#pragma unroll
    for (int r = 0; r < 8; ++r) {
      const float h0v = tanhf(acc0[r]);
      const float h1v = tanhf(acc1[r]);
      const int mrow = m0 + rofs + r;
      hnxt[mrow * HID_ + nc0 + nidx] = f2bf(h0v);
      hnxt[mrow * HID_ + nc1 + nidx] = f2bf(h1v);
      if (t == L_ - 1) {
        out[FINAL_OFS + mrow * HID_ + nc0 + nidx] = h0v;
        out[FINAL_OFS + mrow * HID_ + nc1 + nidx] = h1v;
      }
    }

    // ---- logits_{t-1} = h_{t-1}*W_o + b_o, cols [lncol, lncol+16) ----
    if (t > 0) {
      v8f acc;
#pragma unroll
      for (int r = 0; r < 8; ++r) acc[r] = bias_o;
#pragma unroll 4
      for (int kc = 0; kc < HID_; kc += 32) {
        acc = wmma_bf16(load_frag_row(alrow_l, kc, lane),
                        load_frag_row(bo, kc, lane), acc);
      }
#pragma unroll
      for (int r = 0; r < 8; ++r) {
        const int mrow = lm0 + rofs + r;
        out[(size_t)mrow * (L_ * OUT_) + (size_t)(t - 1) * OUT_ + lncol + nidx] = acc[r];
      }
    }

    grid_barrier(bar, (unsigned)(NWG * (t + 1)));
  }

  // ---- logits_{L-1}: stage h_{L-1} (hbuf[L&1]) and run the last projection ----
  {
    const unsigned short* hlast = hbuf + (size_t)(L_ & 1) * (B_ * HID_);
    for (int i = tid; i < 4096; i += 256) {
      const int row = i >> 6, c8 = (i & 63) * 8;
      async_ld16(&sh_h[row * HPAD + c8], hlast + row * HID_ + c8);
    }
    async_wait_all();
    __syncthreads();

    v8f acc;
#pragma unroll
    for (int r = 0; r < 8; ++r) acc[r] = bias_o;
#pragma unroll 4
    for (int kc = 0; kc < HID_; kc += 32) {
      acc = wmma_bf16(load_frag_row(alrow_l, kc, lane),
                      load_frag_row(bo, kc, lane), acc);
    }
#pragma unroll
    for (int r = 0; r < 8; ++r) {
      const int mrow = lm0 + rofs + r;
      out[(size_t)mrow * (L_ * OUT_) + (size_t)(L_ - 1) * OUT_ + lncol + nidx] = acc[r];
    }
  }
}

extern "C" void kernel_launch(void* const* d_in, const int* in_sizes, int n_in,
                              void* d_out, int out_size, void* d_ws, size_t ws_size,
                              hipStream_t stream) {
  const int*   x         = (const int*)d_in[0];
  const float* hidden    = (const float*)d_in[1];
  const float* embedding = (const float*)d_in[2];
  const float* W_e       = (const float*)d_in[3];
  const float* W_h       = (const float*)d_in[4];
  const float* b_h       = (const float*)d_in[5];
  const float* W_o       = (const float*)d_in[6];
  const float* b_o       = (const float*)d_in[7];
  float* out = (float*)d_out;
  unsigned char* ws = (unsigned char*)d_ws;

  // zero the barrier counter each launch (graph-capturable)
  hipMemsetAsync(ws, 0, 256, stream);

  // prep: 524288 elements total
  charrnn_prep<<<2048, 256, 0, stream>>>(hidden, embedding, W_e, W_h, W_o, ws);

  // persistent recurrence + fused logits
  charrnn_recurrent<<<NWG, 256, 0, stream>>>(x, b_h, b_o, out, ws);
}
